// GATNodeEncoder_13855564497042
// MI455X (gfx1250) — matrix operand order repaired and checked by reading, not compiled, not run on hardware
//
#include <hip/hip_runtime.h>

// ---------------- problem constants (from reference) ----------------
#define N_NODES  100000
#define N_EDGES  1600000
#define E_TOT    (N_EDGES + N_NODES)   // edges + self loops = 1,700,000
#define IN_DIM   128
#define HID      32
#define HEADS    4
#define F1       (HEADS * HID)         // 128
#define OUT_DIM  32
#define NEG_SLOPE 0.2f
#define BN_EPS    1e-5f

typedef __attribute__((ext_vector_type(16))) __bf16 bf16x16;
typedef __attribute__((ext_vector_type(8)))  __bf16 bf16x8;
typedef __attribute__((ext_vector_type(8)))  float  f32x8;

// ---------------- small helpers ----------------
__device__ __forceinline__ float lrelu(float v) { return v > 0.f ? v : NEG_SLOPE * v; }

__device__ __forceinline__ void edge_sd(int e, const int* __restrict__ ei, int& s, int& d) {
  if (e < N_EDGES) { s = ei[e]; d = ei[N_EDGES + e]; }
  else             { s = d = e - N_EDGES; }           // self loop
}

// atomic max for float (works for mixed signs; init memory to -inf)
__device__ __forceinline__ void atomicMaxF(float* a, float v) {
  if (v >= 0.f) atomicMax((int*)a, __float_as_int(v));
  else          atomicMin((unsigned int*)a, __float_as_uint(v));
}

// ---------------- conversion / layout kernels ----------------
__global__ void cvt_bf16_kernel(const float* __restrict__ in, __bf16* __restrict__ out, int n) {
  int t = blockIdx.x * blockDim.x + threadIdx.x;
  if (t < n) out[t] = (__bf16)in[t];
}

// WT[n*K + k] = (bf16) W[k*Nc + n]   (W is [K x Nc] row-major)
__global__ void transpose_cvt_kernel(const float* __restrict__ W, __bf16* __restrict__ WT,
                                     int K, int Nc) {
  int t = blockIdx.x * blockDim.x + threadIdx.x;
  if (t < K * Nc) {
    int k = t / Nc, n = t % Nc;
    WT[n * K + k] = (__bf16)W[t];
  }
}

__global__ void fill_kernel(float* __restrict__ p, float v, int n) {
  int t = blockIdx.x * blockDim.x + threadIdx.x;
  if (t < n) p[t] = v;
}

// ---------------- WMMA GEMM:  C[M x Ncols] = A[M x 128] * B[128 x Ncols] ----------------
// A row-major bf16, BT = B transposed ([Ncols x 128]) bf16, C f32.
// The block's 16x128 A tile is staged once through LDS (padded stride to spread
// LDS banks); each of the NTILES waves computes one 16x16 output tile with
// 4 chained v_wmma_f32_16x16x32_bf16.
#define A_STRIDE 136   // 128 + 8 bf16 pad: row stride 272B -> rows hit different bank groups

template <int NTILES>
__global__ __launch_bounds__(32 * NTILES)
void gemm_bf16_wmma_kernel(const __bf16* __restrict__ A,
                           const __bf16* __restrict__ BT,
                           float* __restrict__ C, int Ncols) {
  __shared__ __bf16 As[16 * A_STRIDE];

  // cooperative stage of the 16x128 A tile (rows are contiguous in global)
  const __bf16* atile = A + (size_t)blockIdx.x * 16 * 128;
  for (int i = threadIdx.x * 8; i < 16 * 128; i += 32 * NTILES * 8) {
    int r = i >> 7, c = i & 127;                       // 128 % 8 == 0: never splits a row
    *(bf16x8*)(As + r * A_STRIDE + c) = *(const bf16x8*)(atile + i);
  }
  __syncthreads();

  const int lane = threadIdx.x & 31;
  const int wave = threadIdx.x >> 5;          // column tile index
  const int hi   = lane >> 4;                 // half-wave select
  const int colg = wave * 16 + (lane & 15);

  f32x8 acc = {};
#pragma unroll
  for (int kk = 0; kk < 128; kk += 32) {
    // A fragment (16x32 bf16): lanes 0-15 hold K {kk..kk+15},
    // lanes 16-31 hold K {kk+16..kk+31}, row = lane&15.
    union { bf16x16 v; bf16x8 h[2]; } a;
    const __bf16* ap = As + (lane & 15) * A_STRIDE + kk + hi * 16;
    a.h[0] = *(const bf16x8*)(ap);
    a.h[1] = *(const bf16x8*)(ap + 8);
    // B fragment (32x16 bf16): lane holds column colg, K range hi*16..hi*16+15 (contiguous in BT)
    bf16x16 b = *(const bf16x16*)(BT + (size_t)colg * 128 + kk + hi * 16);
    acc = __builtin_amdgcn_wmma_f32_16x16x32_bf16(false, a.v, false, b,
                                                  (short)0, acc, false, false);
  }
  // C/D layout: VGPR r -> row = r + 8*hi, col = lane&15
#pragma unroll
  for (int r = 0; r < 8; ++r) {
    int orow = blockIdx.x * 16 + hi * 8 + r;
    C[(size_t)orow * Ncols + colg] = acc[r];
  }
}

// ---------------- attention coefficients: alpha_s/alpha_d per (node, head) ----------------
template <int H, int C>
__global__ void alpha_kernel(const float* __restrict__ h,
                             const float* __restrict__ a_src, const float* __restrict__ a_dst,
                             float* __restrict__ as, float* __restrict__ ad) {
  int t = blockIdx.x * blockDim.x + threadIdx.x;
  if (t >= N_NODES * H) return;
  int n = t / H, hh = t % H;
  const float* hp = h + (size_t)n * (H * C) + hh * C;
  float s = 0.f, d = 0.f;
#pragma unroll
  for (int c = 0; c < C; ++c) {
    float v = hp[c];
    s += v * a_src[hh * C + c];
    d += v * a_dst[hh * C + c];
  }
  as[t] = s; ad[t] = d;
}

// ---------------- edge phase 1: segment max over destinations ----------------
template <int H>
__global__ void edge_max_kernel(const int* __restrict__ ei, const float* __restrict__ as,
                                const float* __restrict__ ad, float* __restrict__ m) {
  int t = blockIdx.x * blockDim.x + threadIdx.x;
  if (t >= E_TOT * H) return;
  int e = t / H, hh = t % H;
  int s, d; edge_sd(e, ei, s, d);
  atomicMaxF(&m[d * H + hh], lrelu(as[s * H + hh] + ad[d * H + hh]));
}

// ---------------- edge phase 2: ex = exp(e - m[dst]); denom += ex ----------------
template <int H>
__global__ void edge_exp_kernel(const int* __restrict__ ei, const float* __restrict__ as,
                                const float* __restrict__ ad, const float* __restrict__ m,
                                float* __restrict__ den, float* __restrict__ ex) {
  int t = blockIdx.x * blockDim.x + threadIdx.x;
  if (t >= E_TOT * H) return;
  int e = t / H, hh = t % H;
  int s, d; edge_sd(e, ei, s, d);
  float ev = lrelu(as[s * H + hh] + ad[d * H + hh]);
  float x = __expf(ev - m[d * H + hh]);
  ex[t] = x;
  atomicAdd(&den[d * H + hh], x);
}

// ---------------- edge phase 3: out[dst] += alpha * h[src]  (float4 gathers) ----------------
template <int H, int C>
__global__ void edge_agg_kernel(const int* __restrict__ ei, const float* __restrict__ ex,
                                const float* __restrict__ den, const float* __restrict__ hfeat,
                                float* __restrict__ out) {
  constexpr int F  = H * C;        // feature width (128 or 32)
  constexpr int FV = F / 4;        // float4 groups per edge
  long long t = (long long)blockIdx.x * blockDim.x + threadIdx.x;
  if (t >= (long long)E_TOT * FV) return;
  int e  = (int)(t / FV);
  int f  = (int)(t % FV) * 4;      // 4 | C, so the group never crosses a head
  int hh = f / C;
  int s, d; edge_sd(e, ei, s, d);
  float alpha = ex[(size_t)e * H + hh] / (den[d * H + hh] + 1e-16f);
  float4 hv = *(const float4*)(hfeat + (size_t)s * F + f);
  float* op = out + (size_t)d * F + f;
  atomicAdd(op + 0, alpha * hv.x);
  atomicAdd(op + 1, alpha * hv.y);
  atomicAdd(op + 2, alpha * hv.z);
  atomicAdd(op + 3, alpha * hv.w);
}

// ---------------- bias + batchnorm(eval) + ELU, emit bf16 activations ----------------
__global__ void bn_elu_kernel(const float* __restrict__ agg, const float* __restrict__ b1,
                              const float* __restrict__ gamma, const float* __restrict__ beta,
                              __bf16* __restrict__ hb) {
  int t = blockIdx.x * blockDim.x + threadIdx.x;
  if (t >= N_NODES * F1) return;
  int f = t & 127;
  float v = agg[t] + b1[f];
  v = v * (gamma[f] * rsqrtf(1.f + BN_EPS)) + beta[f];
  v = v > 0.f ? v : (__expf(v) - 1.f);
  hb[t] = (__bf16)v;
}

// ---------------- final: out = agg2 + b2 ----------------
__global__ void bias_out_kernel(const float* __restrict__ agg, const float* __restrict__ b2,
                                float* __restrict__ out) {
  int t = blockIdx.x * blockDim.x + threadIdx.x;
  if (t >= N_NODES * OUT_DIM) return;
  out[t] = agg[t] + b2[t & 31];
}

// ---------------- workspace layout (bytes, 256-aligned, with aliasing) ----------------
static constexpr size_t XB_OFF   = 0;           // bf16 x; reused as bf16 layer-2 activations
static constexpr size_t W1T_OFF  = 25600000;
static constexpr size_t W2T_OFF  = 25632768;
static constexpr size_t H_OFF    = 25640960;    // f32 h1; reused as h2
static constexpr size_t AS_OFF   = 76840960;
static constexpr size_t AD_OFF   = 78440960;
static constexpr size_t M_OFF    = 80040960;
static constexpr size_t DEN_OFF  = 81640960;
static constexpr size_t EX_OFF   = 83240960;
static constexpr size_t OUT_OFF  = 110440960;   // agg out1; reused as out2

static inline int gridFor(long long n, int b) { return (int)((n + b - 1) / b); }

extern "C" void kernel_launch(void* const* d_in, const int* in_sizes, int n_in,
                              void* d_out, int out_size, void* d_ws, size_t ws_size,
                              hipStream_t stream) {
  (void)in_sizes; (void)n_in; (void)out_size; (void)ws_size;

  const float* x       = (const float*)d_in[0];
  const int*   ei      = (const int*)  d_in[1];
  const float* W1      = (const float*)d_in[2];
  const float* a1_src  = (const float*)d_in[3];
  const float* a1_dst  = (const float*)d_in[4];
  const float* b1      = (const float*)d_in[5];
  const float* bn_g    = (const float*)d_in[6];
  const float* bn_b    = (const float*)d_in[7];
  const float* W2      = (const float*)d_in[8];
  const float* a2_src  = (const float*)d_in[9];
  const float* a2_dst  = (const float*)d_in[10];
  const float* b2      = (const float*)d_in[11];
  float* out = (float*)d_out;

  char* ws = (char*)d_ws;
  __bf16* xb   = (__bf16*)(ws + XB_OFF);
  __bf16* w1t  = (__bf16*)(ws + W1T_OFF);
  __bf16* w2t  = (__bf16*)(ws + W2T_OFF);
  float*  h1   = (float*) (ws + H_OFF);
  float*  as_  = (float*) (ws + AS_OFF);
  float*  ad_  = (float*) (ws + AD_OFF);
  float*  m_   = (float*) (ws + M_OFF);
  float*  den_ = (float*) (ws + DEN_OFF);
  float*  ex_  = (float*) (ws + EX_OFF);
  float*  agg_ = (float*) (ws + OUT_OFF);

  const int B = 256;
  const float NEG_INF = -__builtin_inff();

  // ---- precision conversions / weight transposes
  cvt_bf16_kernel<<<gridFor((long long)N_NODES * IN_DIM, B), B, 0, stream>>>(x, xb, N_NODES * IN_DIM);
  transpose_cvt_kernel<<<gridFor(IN_DIM * F1, B), B, 0, stream>>>(W1, w1t, IN_DIM, F1);
  transpose_cvt_kernel<<<gridFor(F1 * OUT_DIM, B), B, 0, stream>>>(W2, w2t, F1, OUT_DIM);

  // =================== layer 1: GATConv(128 -> 32, heads=4, concat) ===================
  gemm_bf16_wmma_kernel<8><<<N_NODES / 16, 256, 0, stream>>>(xb, w1t, h1, F1);
  alpha_kernel<HEADS, HID><<<gridFor((long long)N_NODES * HEADS, B), B, 0, stream>>>(
      h1, a1_src, a1_dst, as_, ad_);
  fill_kernel<<<gridFor((long long)N_NODES * HEADS, B), B, 0, stream>>>(m_,   NEG_INF, N_NODES * HEADS);
  fill_kernel<<<gridFor((long long)N_NODES * HEADS, B), B, 0, stream>>>(den_, 0.f,     N_NODES * HEADS);
  fill_kernel<<<gridFor((long long)N_NODES * F1,    B), B, 0, stream>>>(agg_, 0.f,     N_NODES * F1);
  edge_max_kernel<HEADS><<<gridFor((long long)E_TOT * HEADS, B), B, 0, stream>>>(ei, as_, ad_, m_);
  edge_exp_kernel<HEADS><<<gridFor((long long)E_TOT * HEADS, B), B, 0, stream>>>(ei, as_, ad_, m_, den_, ex_);
  edge_agg_kernel<HEADS, HID><<<gridFor((long long)E_TOT * (F1 / 4), B), B, 0, stream>>>(
      ei, ex_, den_, h1, agg_);

  // ---- +b1, BatchNorm(eval) + ELU -> bf16 activations (into xb region)
  bn_elu_kernel<<<gridFor((long long)N_NODES * F1, B), B, 0, stream>>>(agg_, b1, bn_g, bn_b, xb);

  // =================== layer 2: GATConv(128 -> 32, heads=1, mean) ===================
  gemm_bf16_wmma_kernel<2><<<N_NODES / 16, 64, 0, stream>>>(xb, w2t, h1, OUT_DIM);  // h1 now holds h2
  alpha_kernel<1, OUT_DIM><<<gridFor((long long)N_NODES, B), B, 0, stream>>>(h1, a2_src, a2_dst, as_, ad_);
  fill_kernel<<<gridFor((long long)N_NODES, B), B, 0, stream>>>(m_,   NEG_INF, N_NODES);
  fill_kernel<<<gridFor((long long)N_NODES, B), B, 0, stream>>>(den_, 0.f,     N_NODES);
  fill_kernel<<<gridFor((long long)N_NODES * OUT_DIM, B), B, 0, stream>>>(agg_, 0.f, N_NODES * OUT_DIM);
  edge_max_kernel<1><<<gridFor((long long)E_TOT, B), B, 0, stream>>>(ei, as_, ad_, m_);
  edge_exp_kernel<1><<<gridFor((long long)E_TOT, B), B, 0, stream>>>(ei, as_, ad_, m_, den_, ex_);
  edge_agg_kernel<1, OUT_DIM><<<gridFor((long long)E_TOT * (OUT_DIM / 4), B), B, 0, stream>>>(
      ei, ex_, den_, h1, agg_);

  bias_out_kernel<<<gridFor((long long)N_NODES * OUT_DIM, B), B, 0, stream>>>(agg_, b2, out);
}
